// LSTMTagger_7000796692518
// MI455X (gfx1250) — compile-verified, matrix operand order checked by reading, not run on hardware
//
#include <hip/hip_runtime.h>
#include <hip/hip_bf16.h>
#include <math.h>

// ---------------- problem constants ----------------
#define S_LEN   4096
#define EMB_D   1024
#define HID_D   1024
#define FOURH   4096
#define TAGS_D  50

// recurrence persistent-kernel config
#define RNN_WGS      64            // workgroups in persistent grid
#define RNN_THREADS  256           // 64 rows x 4 K-partitions
#define ROWS_PER_WG  64            // 4 gates x 16 hidden units
#define UNITS_PER_WG 16
#define PADW         1032          // bf16 row stride in LDS: 2064B = 129*16B aligned, 516 dw == 4 mod 64 banks

// GEMM config
#define GEMM_THREADS 256           // 8 waves -> 128-row M tile
#define BPAD         1040          // bf16 row stride: 2080B = 65*32B aligned, 520 dw == 8 mod 64 banks

typedef __attribute__((ext_vector_type(16))) __bf16 v16bf;
typedef __attribute__((ext_vector_type(8)))  float  v8f;

// ---------------- CDNA5 async global->LDS (guarded) ----------------
#if __has_builtin(__builtin_amdgcn_global_load_async_to_lds_b128)
#define HAVE_ASYNC_LDS 1
#endif

// Builtin signature (from compiler diagnostic): pointee is GCC-vector
// 'int __attribute__((vector_size(16)))', AS1 source / AS3 dest.
typedef int v4i_gcc __attribute__((vector_size(16)));
typedef __attribute__((address_space(1))) v4i_gcc* as1_v4i;
typedef __attribute__((address_space(3))) v4i_gcc* as3_v4i;

__device__ __forceinline__ void gld_async_b128(const void* g, void* l) {
#ifdef HAVE_ASYNC_LDS
  __builtin_amdgcn_global_load_async_to_lds_b128((as1_v4i)g, (as3_v4i)l, 0, 0);
#else
  // fallback: regular load + LDS store (16B)
  *(uint4*)l = *(const uint4*)g;
#endif
}

__device__ __forceinline__ void wait_async0() {
#ifdef HAVE_ASYNC_LDS
#if __has_builtin(__builtin_amdgcn_s_wait_asynccnt)
  __builtin_amdgcn_s_wait_asynccnt(0);
#else
  asm volatile("s_wait_asynccnt 0" ::: "memory");
#endif
#endif
}

// ---------------- workspace layout (bytes) ----------------
// sync(256) | x_bf16 8MB | Wih_bf16 8MB | Whh_bf16 8MB | x_proj 64MB | lstm_out 16MB | h_buf 4KB
static constexpr size_t OFF_SYNC  = 0;
static constexpr size_t OFF_X     = 256;
static constexpr size_t SZ_X      = (size_t)S_LEN * EMB_D * 2;
static constexpr size_t OFF_WIH   = OFF_X + SZ_X;
static constexpr size_t SZ_W      = (size_t)FOURH * EMB_D * 2;
static constexpr size_t OFF_WHH   = OFF_WIH + SZ_W;
static constexpr size_t OFF_XPROJ = OFF_WHH + SZ_W;
static constexpr size_t SZ_XPROJ  = (size_t)S_LEN * FOURH * 4;
static constexpr size_t OFF_LOUT  = OFF_XPROJ + SZ_XPROJ;
static constexpr size_t SZ_LOUT   = (size_t)S_LEN * HID_D * 4;
static constexpr size_t OFF_HBUF  = OFF_LOUT + SZ_LOUT;

// ---------------- kernels ----------------

__global__ void embed_gather_cast(const int* __restrict__ sent,
                                  const float* __restrict__ emb,
                                  __hip_bfloat16* __restrict__ xout) {
  int i = blockIdx.x * blockDim.x + threadIdx.x;
  if (i >= S_LEN * EMB_D) return;
  int s = i >> 10, e = i & 1023;
  xout[i] = __float2bfloat16(emb[(size_t)sent[s] * EMB_D + e]);
}

__global__ void cast_f32_bf16(const float* __restrict__ in,
                              __hip_bfloat16* __restrict__ out, int n) {
  int i = blockIdx.x * blockDim.x + threadIdx.x;
  if (i < n) out[i] = __float2bfloat16(in[i]);
}

// x_proj[S,4H] = X(bf16) @ W_ih^T(bf16) + (b_ih + b_hh), WMMA bf16 16x16x32.
// Block = 256 threads (8 waves). The 64-column W tile (full K) is staged once
// into LDS with async global->LDS loads and reused by all 8 M-subtiles.
__global__ __launch_bounds__(GEMM_THREADS)
void xproj_gemm_wmma(const __hip_bfloat16* __restrict__ X,
                     const __hip_bfloat16* __restrict__ W,     // [4H, E] row-major == B matrix
                     const float* __restrict__ b_ih,
                     const float* __restrict__ b_hh,
                     float* __restrict__ xproj) {
  __shared__ __hip_bfloat16 sB[64 * BPAD];   // ~130 KB

  const int tid   = threadIdx.x;
  const int lane  = tid & 31;
  const int wave  = tid >> 5;
  const int mbase = blockIdx.y * 128 + wave * 16;
  const int nblk  = blockIdx.x * 64;
  const int l16   = lane & 15;
  const int koff  = (lane >> 4) << 4;     // contiguous k order used here: lanes 16-31 hold K+16

  // stage B tile: 64 rows x 1024 bf16 in 16B chunks (8192 chunks, 32/thread)
  for (int c = tid; c < 64 * 128; c += GEMM_THREADS) {
    int r = c >> 7, cs = c & 127;
    gld_async_b128(W + (size_t)(nblk + r) * EMB_D + cs * 8,
                   &sB[r * BPAD + cs * 8]);
  }
  wait_async0();
  __syncthreads();

  v8f acc[4];
  #pragma unroll
  for (int nt = 0; nt < 4; ++nt) acc[nt] = (v8f){0.f,0.f,0.f,0.f,0.f,0.f,0.f,0.f};

  const __hip_bfloat16* aptr = X + (size_t)(mbase + l16) * EMB_D + koff;

  for (int k0 = 0; k0 < EMB_D; k0 += 32) {
    __builtin_prefetch(aptr + k0 + 128, 0, 1);                 // global_prefetch_b8
    v16bf a = *reinterpret_cast<const v16bf*>(aptr + k0);
    #pragma unroll
    for (int nt = 0; nt < 4; ++nt) {
      const __hip_bfloat16* bp = &sB[(nt * 16 + l16) * BPAD + koff + k0];
      v16bf b = *reinterpret_cast<const v16bf*>(bp);           // ds_load_b128 x2
      acc[nt] = __builtin_amdgcn_wmma_f32_16x16x32_bf16(
          false, a, false, b, (short)0, acc[nt], false, false);
    }
  }

  // D layout: VGPR r -> M = mbase + r + 8*(lane>>4), N = nbase + (lane&15)
  const int rbase = mbase + ((lane >> 4) << 3);
  #pragma unroll
  for (int nt = 0; nt < 4; ++nt) {
    const int col  = nblk + nt * 16 + l16;
    const float bs = b_ih[col] + b_hh[col];
    #pragma unroll
    for (int r = 0; r < 8; ++r)
      xproj[(size_t)(rbase + r) * FOURH + col] = acc[nt][r] + bs;
  }
}

__device__ __forceinline__ float sigmoidf_(float x) {
  return 1.0f / (1.0f + __expf(-x));
}

__device__ void grid_barrier(int* cnt, volatile int* gen, int nwg) {
  __syncthreads();
  if (threadIdx.x == 0) {
    __threadfence();
    int g = *gen;
    if (atomicAdd(cnt, 1) == nwg - 1) {
      *cnt = 0;
      __threadfence();
      atomicAdd((int*)gen, 1);
    } else {
      while (*gen == g) { __builtin_amdgcn_s_sleep(1); }
    }
    __threadfence();
  }
  __syncthreads();
}

// Persistent recurrence: 64 WGs, each owns 16 hidden units (64 gate rows).
// W_hh slice (bf16, padded) lives in LDS for all 4096 steps; h is exchanged
// through L2 with a software grid barrier each step.
__global__ __launch_bounds__(RNN_THREADS)
void lstm_recurrence(const __hip_bfloat16* __restrict__ Whh,  // [4H, H] bf16 row-major
                     const float* __restrict__ xproj,          // [S, 4H]
                     const float* __restrict__ h0,
                     const float* __restrict__ c0,
                     float* __restrict__ lstm_out,              // [S, H]
                     float* __restrict__ h_buf,                 // [H]
                     int* __restrict__ sync_mem) {
  __shared__ __hip_bfloat16 sW[ROWS_PER_WG * PADW];  // ~129 KB
  __shared__ float sh[HID_D];
  __shared__ float sg[ROWS_PER_WG];

  const int tid  = threadIdx.x;
  const int wg   = blockIdx.x;
  const int lr   = tid >> 2;        // local gate-row 0..63
  const int part = tid & 3;         // K partition 0..3
  const int gate = lr >> 4;         // 0=i 1=f 2=g 3=o
  const int ul   = lr & 15;         // unit within WG
  const int grow = gate * HID_D + wg * UNITS_PER_WG + ul;   // global W_hh row

  // stage this WG's W_hh slice into LDS once (async, 16B chunks)
  for (int c = tid; c < ROWS_PER_WG * 128; c += RNN_THREADS) {
    int r = c >> 7, cs = c & 127;
    int gr = (r >> 4) * HID_D + wg * UNITS_PER_WG + (r & 15);
    gld_async_b128(Whh + (size_t)gr * HID_D + cs * 8,
                   &sW[r * PADW + cs * 8]);
  }
  // h_{-1} = h0
  for (int j = tid; j < HID_D; j += RNN_THREADS) sh[j] = h0[j];
  float c_reg = (tid < UNITS_PER_WG) ? c0[wg * UNITS_PER_WG + tid] : 0.f;
  wait_async0();
  __syncthreads();

  int* cnt = sync_mem;
  volatile int* gen = sync_mem + 1;

  for (int t = 0; t < S_LEN; ++t) {
    // dot(h, W_hh[grow,:]) split over 4 lanes, K interleaved 4-way
    float acc = 0.f;
    const __hip_bfloat16* wrow = &sW[lr * PADW];
    #pragma unroll 16
    for (int i = 0; i < HID_D / 4; ++i) {
      int k = (i << 2) | part;
      acc = fmaf(__bfloat162float(wrow[k]), sh[k], acc);
    }
    acc += __shfl_xor(acc, 1, 32);
    acc += __shfl_xor(acc, 2, 32);
    if (part == 0)
      sg[lr] = acc + xproj[(size_t)t * FOURH + grow];
    __syncthreads();

    if (tid < UNITS_PER_WG) {
      float gi = sigmoidf_(sg[tid]);
      float gf = sigmoidf_(sg[UNITS_PER_WG + tid]);
      float gg = tanhf   (sg[2 * UNITS_PER_WG + tid]);
      float go = sigmoidf_(sg[3 * UNITS_PER_WG + tid]);
      c_reg = gf * c_reg + gi * gg;
      float h = go * tanhf(c_reg);
      int u = wg * UNITS_PER_WG + tid;
      h_buf[u] = h;
      lstm_out[(size_t)t * HID_D + u] = h;
    }
    grid_barrier(cnt, gen, RNN_WGS);          // all h slices visible device-wide
    for (int j = tid; j < HID_D; j += RNN_THREADS) sh[j] = h_buf[j];
    __syncthreads();
  }
}

// tag_space = lstm_out @ W_out^T + b_out, then row log-softmax. One block per row.
__global__ __launch_bounds__(64)
void tags_logsoftmax(const float* __restrict__ lstm_out,
                     const float* __restrict__ Wout,   // [TAGS, H]
                     const float* __restrict__ bout,
                     float* __restrict__ out) {        // [S, TAGS]
  __shared__ float sh[HID_D];
  __shared__ float st[64];
  const int row = blockIdx.x;
  const int tid = threadIdx.x;
  for (int j = tid; j < HID_D; j += 64) sh[j] = lstm_out[(size_t)row * HID_D + j];
  __syncthreads();
  if (tid < TAGS_D) {
    const float* w = Wout + (size_t)tid * HID_D;
    float a = bout[tid];
    #pragma unroll 8
    for (int k = 0; k < HID_D; ++k) a = fmaf(w[k], sh[k], a);
    st[tid] = a;
  }
  __syncthreads();
  if (tid < TAGS_D) {
    float m = -1e30f;
    for (int j = 0; j < TAGS_D; ++j) m = fmaxf(m, st[j]);
    float s = 0.f;
    for (int j = 0; j < TAGS_D; ++j) s += __expf(st[j] - m);
    out[(size_t)row * TAGS_D + tid] = st[tid] - m - __logf(s);
  }
}

// ---------------- host launcher ----------------
extern "C" void kernel_launch(void* const* d_in, const int* in_sizes, int n_in,
                              void* d_out, int out_size, void* d_ws, size_t ws_size,
                              hipStream_t stream) {
  const int*   sentence = (const int*)  d_in[0];
  const float* embed    = (const float*)d_in[1];
  const float* W_ih     = (const float*)d_in[2];
  const float* W_hh     = (const float*)d_in[3];
  const float* b_ih     = (const float*)d_in[4];
  const float* b_hh     = (const float*)d_in[5];
  const float* W_out    = (const float*)d_in[6];
  const float* b_out    = (const float*)d_in[7];
  const float* h0       = (const float*)d_in[8];
  const float* c0       = (const float*)d_in[9];
  float* out = (float*)d_out;

  char* ws = (char*)d_ws;
  int*            syncm   = (int*)(ws + OFF_SYNC);
  __hip_bfloat16* x_bf    = (__hip_bfloat16*)(ws + OFF_X);
  __hip_bfloat16* wih_bf  = (__hip_bfloat16*)(ws + OFF_WIH);
  __hip_bfloat16* whh_bf  = (__hip_bfloat16*)(ws + OFF_WHH);
  float*          xproj   = (float*)(ws + OFF_XPROJ);
  float*          lstm_o  = (float*)(ws + OFF_LOUT);
  float*          h_buf   = (float*)(ws + OFF_HBUF);

  (void)hipMemsetAsync(syncm, 0, 256, stream);

  // embedding gather + bf16 cast
  {
    int n = S_LEN * EMB_D;
    embed_gather_cast<<<(n + 255) / 256, 256, 0, stream>>>(sentence, embed, x_bf);
  }
  // weight casts
  {
    int n = FOURH * EMB_D;
    cast_f32_bf16<<<(n + 255) / 256, 256, 0, stream>>>(W_ih, wih_bf, n);
    cast_f32_bf16<<<(n + 255) / 256, 256, 0, stream>>>(W_hh, whh_bf, n);
  }
  // x_proj GEMM (WMMA bf16, LDS-staged B tile)
  {
    dim3 grid(FOURH / 64, S_LEN / 128);
    xproj_gemm_wmma<<<grid, GEMM_THREADS, 0, stream>>>(x_bf, wih_bf, b_ih, b_hh, xproj);
  }
  // persistent LSTM recurrence
  lstm_recurrence<<<RNN_WGS, RNN_THREADS, 0, stream>>>(
      whh_bf, xproj, h0, c0, lstm_o, h_buf, syncm);
  // output projection + log-softmax
  tags_logsoftmax<<<S_LEN, 64, 0, stream>>>(lstm_o, W_out, b_out, out);
}
